// Encoder_26474178413106
// MI455X (gfx1250) — matrix operand order repaired and checked by reading, not compiled, not run on hardware
//
#include <hip/hip_runtime.h>
#include <hip/hip_bf16.h>

// ---------------------------------------------------------------------------
// Transformer encoder fwd on MI455X (gfx1250):
//  * all GEMMs via v_wmma_f32_16x16x32_f16 (wave32, f32 accumulate)
//  * GEMM tiles double-buffered in LDS, fed by the CDNA5 async paths:
//      A-tile: global_load_async_to_lds_b128 (ASYNCcnt)
//      B-tile: TENSOR_LOAD_TO_LDS via the Tensor Data Mover (TENSORcnt)
//  * fused attention keeps the 32x1024 fp32 score panel in LDS
//    (128 KB -- CDNA5 WGP has 320 KB LDS -> 2 blocks/WGP)
// ---------------------------------------------------------------------------

typedef __attribute__((ext_vector_type(16))) _Float16 v16h;
typedef __attribute__((ext_vector_type(8)))  _Float16 v8h;
typedef __attribute__((ext_vector_type(8)))  float    v8f;
typedef __attribute__((ext_vector_type(4)))  unsigned v4u;
typedef __attribute__((ext_vector_type(8)))  int      v8i;
typedef __attribute__((ext_vector_type(4)))  int      v4i;

#define BATCH 8
#define SEQ   1024
#define DMODEL 1024
#define NHEAD 16
#define HDIM  64
#define DFF   4096
#define NLAYER 4
#define MROWS (BATCH * SEQ)   // 8192
#define EPS   1e-3f

#if defined(__has_builtin)
#if __has_builtin(__builtin_amdgcn_tensor_load_to_lds)
#define HAVE_TDM 1
#endif
#endif

// ---------------- WMMA / async helpers ------------------------------------

__device__ __forceinline__ v8f wmma_f16(v16h a, v16h b, v8f c) {
  // D = A(16x32 f16) * B(32x16 f16) + C(16x16 f32)
  return __builtin_amdgcn_wmma_f32_16x16x32_f16(
      /*neg_a=*/false, a, /*neg_b=*/false, b,
      /*c_mod=*/(short)0, c, /*reuse_a=*/false, /*reuse_b=*/false);
}

// Build a v16h fragment from two aligned 8-half chunks.
__device__ __forceinline__ v16h frag_pair(const _Float16* p0, const _Float16* p1) {
  v8h lo = *(const v8h*)p0;
  v8h hq = *(const v8h*)p1;
  v16h a;
#pragma unroll
  for (int e = 0; e < 8; ++e) { a[e] = lo[e]; a[e + 8] = hq[e]; }
  return a;
}

// Per-lane async copy: 16 bytes global -> LDS, tracked by ASYNCcnt.
// (inline asm: portable across ROCm-7.2 / amdgpu-toolchain builtin skews)
__device__ __forceinline__ void async_ld_b128(unsigned lds_off, const _Float16* g) {
  asm volatile("global_load_async_to_lds_b128 %0, %1, off"
               :: "v"(lds_off), "v"(g) : "memory");
}
__device__ __forceinline__ void wait_asynccnt0() {
  asm volatile("s_wait_asynccnt 0x0" ::: "memory");
}

#ifdef HAVE_TDM
// TDM 2-D tile load: rows x rowlen (2-byte elements) from a row-major tensor
// with row stride `stride_elems` into contiguous LDS at `lds_off`.
__device__ __forceinline__ void tdm_load_2d(unsigned lds_off, const void* gaddr,
                                            unsigned rows, unsigned rowlen,
                                            unsigned tensor_w, unsigned tensor_h,
                                            unsigned long long stride_elems) {
  unsigned long long ga = (unsigned long long)gaddr;
  v4u g0;
  g0.x = 1u;                                            // count=1 (valid user D#)
  g0.y = lds_off;                                       // lds_addr
  g0.z = (unsigned)(ga & 0xffffffffu);                  // global_addr[31:0]
  g0.w = (unsigned)((ga >> 32) & 0x01ffffffu) | (2u << 30);  // [56:32] | type=2
  v8i g1;
  g1[0] = (int)(1u << 16);                              // data_size=1 (2 bytes)
  g1[1] = (int)((tensor_w & 0xffffu) << 16);            // tensor_dim0[15:0]
  g1[2] = (int)((tensor_w >> 16) | ((tensor_h & 0xffffu) << 16));
  g1[3] = (int)((tensor_h >> 16) | ((rowlen & 0xffffu) << 16));  // tile_dim0
  g1[4] = (int)(rows & 0xffffu);                        // tile_dim1 (tile_dim2=0)
  g1[5] = (int)(stride_elems & 0xffffffffu);            // tensor_dim0_stride lo
  g1[6] = (int)((stride_elems >> 32) & 0xffffu);        // stride hi (dim1_stride=0)
  g1[7] = 0;
  v4i z4 = {0, 0, 0, 0};                                // groups 2/3: NULL (2-D)
  v8i z8 = {0, 0, 0, 0, 0, 0, 0, 0};
  // 6-arg form (amdgpu-toolchain / clang-23 signature)
  __builtin_amdgcn_tensor_load_to_lds(g0, g1, z4, z4, z8, 0);
}
#endif

// ---------------- weight repack / convert kernels ------------------------

// wq/wk/wv [L,H,D,HD] f32 -> wqkv_h [L][D][3*1024] f16, n = which*1024+h*64+e
__global__ void repack_qkv_w(const float* __restrict__ wq,
                             const float* __restrict__ wk,
                             const float* __restrict__ wv,
                             _Float16* __restrict__ dst, int total) {
  int i = blockIdx.x * blockDim.x + threadIdx.x;
  if (i >= total) return;
  const int perL = DMODEL * 3 * DMODEL;
  int l = i / perL;
  int rem = i - l * perL;
  int d = rem / (3 * DMODEL);
  int n = rem - d * (3 * DMODEL);
  int which = n >> 10;
  int h = (n >> 6) & (NHEAD - 1);
  int e = n & (HDIM - 1);
  const float* src = (which == 0) ? wq : (which == 1) ? wk : wv;
  size_t sidx = (((size_t)l * NHEAD + h) * DMODEL + d) * HDIM + e;
  dst[i] = (_Float16)src[sidx];
}

// bq/bk/bv [L,H,HD] f32 -> bqkv [L][3072] f32
__global__ void repack_qkv_b(const float* __restrict__ bq,
                             const float* __restrict__ bk,
                             const float* __restrict__ bv,
                             float* __restrict__ dst, int total) {
  int i = blockIdx.x * blockDim.x + threadIdx.x;
  if (i >= total) return;
  int l = i / (3 * DMODEL);
  int n = i - l * (3 * DMODEL);
  int which = n >> 10;
  int h = (n >> 6) & (NHEAD - 1);
  int e = n & (HDIM - 1);
  const float* src = (which == 0) ? bq : (which == 1) ? bk : bv;
  dst[i] = src[((size_t)l * NHEAD + h) * HDIM + e];
}

__global__ void cvt_f32_f16(const float* __restrict__ src,
                            _Float16* __restrict__ dst, int total) {
  int i = blockIdx.x * blockDim.x + threadIdx.x;
  if (i < total) dst[i] = (_Float16)src[i];
}

// ---------------- embedding + positional encoding ------------------------

__global__ void embed_kernel(const int* __restrict__ seq,
                             const float* __restrict__ emb,
                             const float* __restrict__ pes,
                             float* __restrict__ x, _Float16* __restrict__ xh) {
  int i = blockIdx.x * blockDim.x + threadIdx.x;   // over B*S*D
  int d = i & (DMODEL - 1);
  int bs = i >> 10;
  int s = bs & (SEQ - 1);
  int tok = seq[bs];
  float v = emb[(size_t)tok * DMODEL + d] + pes[(size_t)s * DMODEL + d];
  x[i] = v;
  xh[i] = (_Float16)v;
}

// ---------------- generic WMMA GEMM --------------------------------------
// C[M,N] = A_h[M,K] * B_h[K,N] (+bias). 256 threads = 8 wave32, block tile
// 64x128, each wave computes a 32x32 tile (2x2 WMMA, v8f accumulators).
// LDS tiles are double-buffered and filled by CDNA5 async paths:
//   A-tile (64x32, [m][k]) : per-lane global_load_async_to_lds_b128
//   B-tile (32x128, [k][n]): TDM tensor_load_to_lds issued by wave 0
// One barrier per K-step: waits(asynccnt/tensorcnt) -> barrier -> issue next
// tile into the other buffer -> WMMA on current buffer.
// MODE 0: scatter f16 -> q/k/v [B,H,S,HD]   (N = 3072)
// MODE 1: f32 out[m*N+n] = acc + bias[n]
// MODE 2: f16 out[m*N+n] = relu(acc + bias[n])

#define GBM 64
#define GBN 128
#define GBK 32

template <int MODE>
__global__ void gemm_wmma(const _Float16* __restrict__ A,
                          const _Float16* __restrict__ Bw,
                          const float* __restrict__ bias,
                          float* __restrict__ outF,
                          _Float16* __restrict__ outH,
                          _Float16* __restrict__ qo,
                          _Float16* __restrict__ ko,
                          _Float16* __restrict__ vo,
                          int N, int K) {
  __shared__ __align__(16) _Float16 As[2][GBM * GBK];  // 2 x 4 KB, [m][k]
  __shared__ __align__(16) _Float16 Bs[2][GBK * GBN];  // 2 x 8 KB, [k][n]

  const int t = threadIdx.x;
  const int lane = t & 31, wid = t >> 5;
  const int waveM = wid >> 2, waveN = wid & 3;   // 2 x 4 waves
  const int m0 = blockIdx.y * GBM, n0 = blockIdx.x * GBN;
  const int r = lane & 15, hi = lane >> 4;

  const int arow = t >> 2, acol = (t & 3) * 8;   // A-tile: 16B per thread
#ifndef HAVE_TDM
  const int bkk = t >> 3, bnb = (t & 7) * 16;    // B-tile fallback: 32B per thread
#endif

  auto issue_tiles = [&](int buf, int k0) {
    async_ld_b128((unsigned)(size_t)&As[buf][arow * GBK + acol],
                  &A[(size_t)(m0 + arow) * K + k0 + acol]);
#ifdef HAVE_TDM
    if (wid == 0) {
      tdm_load_2d((unsigned)(size_t)&Bs[buf][0],
                  &Bw[(size_t)k0 * N + n0],
                  /*rows=*/GBK, /*rowlen=*/GBN,
                  /*tensor_w=*/(unsigned)N, /*tensor_h=*/(unsigned)K,
                  /*stride=*/(unsigned long long)N);
    }
#else
    async_ld_b128((unsigned)(size_t)&Bs[buf][bkk * GBN + bnb],
                  &Bw[(size_t)(k0 + bkk) * N + n0 + bnb]);
    async_ld_b128((unsigned)(size_t)&Bs[buf][bkk * GBN + bnb + 8],
                  &Bw[(size_t)(k0 + bkk) * N + n0 + bnb + 8]);
#endif
  };

  v8f acc[2][2] = {};
  const int KT = K / GBK;
  issue_tiles(0, 0);

  for (int kt = 0; kt < KT; ++kt) {
    const int cur = kt & 1;
    wait_asynccnt0();
#ifdef HAVE_TDM
    if (wid == 0) __builtin_amdgcn_s_wait_tensorcnt((short)0);
#endif
    __syncthreads();                 // tile `cur` ready; buf cur^1 free
    if (kt + 1 < KT) issue_tiles(cur ^ 1, (kt + 1) * GBK);

    // A frags: lane = row m=r; half i -> k = i + hi*16
    v16h afr[2];
#pragma unroll
    for (int i = 0; i < 2; ++i) {
      int mr = waveM * 32 + i * 16 + r;
      afr[i] = frag_pair(&As[cur][mr * GBK + hi * 16],
                         &As[cur][mr * GBK + hi * 16 + 8]);
    }
    // B frags from [k][n] LDS: lane = col n; half i -> k = i + hi*16
    v16h bfr[2];
#pragma unroll
    for (int j = 0; j < 2; ++j) {
      int nc = waveN * 32 + j * 16 + r;
      v16h b;
#pragma unroll
      for (int i2 = 0; i2 < 16; ++i2)
        b[i2] = Bs[cur][(i2 + hi * 16) * GBN + nc];
      bfr[j] = b;
    }
#pragma unroll
    for (int i = 0; i < 2; ++i)
#pragma unroll
      for (int j = 0; j < 2; ++j)
        acc[i][j] = wmma_f16(afr[i], bfr[j], acc[i][j]);
  }

  // epilogue: D layout -> element (vj, lane): m_local = vj + hi*8, n_local = r
#pragma unroll
  for (int i = 0; i < 2; ++i) {
#pragma unroll
    for (int j = 0; j < 2; ++j) {
      int mbase = m0 + waveM * 32 + i * 16 + hi * 8;
      int n = n0 + waveN * 32 + j * 16 + r;
      float bv = bias ? bias[n] : 0.f;
#pragma unroll
      for (int vj = 0; vj < 8; ++vj) {
        int m = mbase + vj;
        float val = acc[i][j][vj] + bv;
        if (MODE == 0) {
          int which = n >> 10;
          int h = (n >> 6) & (NHEAD - 1);
          int e = n & (HDIM - 1);
          int b = m >> 10, s = m & (SEQ - 1);
          _Float16* dst = (which == 0) ? qo : (which == 1) ? ko : vo;
          dst[(((size_t)b * NHEAD + h) * SEQ + s) * HDIM + e] = (_Float16)val;
        } else if (MODE == 1) {
          outF[(size_t)m * N + n] = val;
        } else {
          outH[(size_t)m * N + n] = (_Float16)(val > 0.f ? val : 0.f);
        }
      }
    }
  }
}

// ---------------- fused attention -----------------------------------------
// One block = one (b,h, 32-row Q tile). The full 32x1024 fp32 score panel
// lives in LDS (128 KB -- CDNA5 WGP has 320 KB). Stages:
//   scores = Q K^T            -> LDS   (WMMA, K streamed from global)
//   row softmax in LDS                 (wave32 shuffle reductions)
//   out = P V                          (WMMA, A-frag converted f32->f16)

__global__ void attn_fused(const _Float16* __restrict__ q,
                           const _Float16* __restrict__ k,
                           const _Float16* __restrict__ v,
                           _Float16* __restrict__ heads) {
  extern __shared__ float sc[];                  // 32 * 1024 floats = 128 KB
  const int blk = blockIdx.x;
  const int qt = blk & 31;                       // S/32 = 32 q-tiles
  const int bh = blk >> 5;                       // b*H + h
  const int b = bh >> 4, h = bh & (NHEAD - 1);
  const int q0 = qt * 32;
  const _Float16* qbase = q + (size_t)bh * SEQ * HDIM;
  const _Float16* kbase = k + (size_t)bh * SEQ * HDIM;
  const _Float16* vbase = v + (size_t)bh * SEQ * HDIM;
  const int t = threadIdx.x, lane = t & 31, wid = t >> 5;
  const int r = lane & 15, hi = lane >> 4;

  // Preload Q A-fragments: [row-tile 0..1][k-chunk 0..1] over HD=64
  v16h aq[2][2];
#pragma unroll
  for (int rt = 0; rt < 2; ++rt)
#pragma unroll
    for (int kc = 0; kc < 2; ++kc) {
      const _Float16* p = &qbase[(size_t)(q0 + rt * 16 + r) * HDIM + kc * 32 + hi * 16];
      aq[rt][kc] = frag_pair(p, p + 8);
    }

  // ---- scores: 128 (2x64) 16x16 tiles split across 8 waves ----
  for (int tile = wid; tile < 128; tile += 8) {
    int rt = tile & 1, kt = tile >> 1;
    v8f a = {};
#pragma unroll
    for (int kc = 0; kc < 2; ++kc) {
      // B frag = K^T: lane -> key = kt*16 + r ; halfs i -> kk = kc*32 + i + hi*16
      const _Float16* p = &kbase[(size_t)(kt * 16 + r) * HDIM + kc * 32 + hi * 16];
      v16h bfr = frag_pair(p, p + 8);
      a = wmma_f16(aq[rt][kc], bfr, a);
    }
#pragma unroll
    for (int vj = 0; vj < 8; ++vj)
      sc[(rt * 16 + hi * 8 + vj) * SEQ + kt * 16 + r] = a[vj];
  }
  __syncthreads();

  // ---- softmax: 4 rows per wave, lanes stride the 1024 columns ----
  const float scale = 0.125f;  // 1/sqrt(64)
#pragma unroll
  for (int rr = 0; rr < 4; ++rr) {
    float* rp = &sc[(wid * 4 + rr) * SEQ];
    float mx = -3.4e38f;
    for (int c = lane; c < SEQ; c += 32) mx = fmaxf(mx, rp[c] * scale);
#pragma unroll
    for (int o = 16; o; o >>= 1) mx = fmaxf(mx, __shfl_xor(mx, o, 32));
    float sum = 0.f;
    for (int c = lane; c < SEQ; c += 32) {
      float e = __expf(rp[c] * scale - mx);
      rp[c] = e;
      sum += e;
    }
#pragma unroll
    for (int o = 16; o; o >>= 1) sum += __shfl_xor(sum, o, 32);
    float inv = 1.f / sum;
    for (int c = lane; c < SEQ; c += 32) rp[c] *= inv;
  }
  __syncthreads();

  // ---- out = P @ V : 8 output tiles (2 row x 4 hd) -> one per wave ----
  const int rt = wid >> 2, hc = wid & 3;
  v8f acco = {};
  for (int kc = 0; kc < SEQ / 32; ++kc) {
    // A frag from LDS scores, f32 -> f16 on load
    const float* rp = &sc[(size_t)(rt * 16 + r) * SEQ + kc * 32 + hi * 16];
    v16h a;
#pragma unroll
    for (int e = 0; e < 8; ++e) {
      a[e]     = (_Float16)rp[e];
      a[e + 8] = (_Float16)rp[8 + e];
    }
    // B frag = V: lane -> hd col hc*16 + r ; halfs i -> key = kc*32 + i + hi*16
    v16h bfr;
#pragma unroll
    for (int i2 = 0; i2 < 16; ++i2)
      bfr[i2] = vbase[(size_t)(kc * 32 + i2 + hi * 16) * HDIM + hc * 16 + r];
    acco = wmma_f16(a, bfr, acco);
  }
  // store heads as [B,S,H,HD] so the out-proj GEMM sees contiguous D rows
#pragma unroll
  for (int vj = 0; vj < 8; ++vj) {
    int s = q0 + rt * 16 + hi * 8 + vj;
    heads[(((size_t)b * SEQ + s) * NHEAD + h) * HDIM + hc * 16 + r] =
        (_Float16)acco[vj];
  }
}

// ---------------- residual + inference BatchNorm --------------------------

__global__ void bn_residual(const float* __restrict__ x,
                            const float* __restrict__ tadd,
                            const float* __restrict__ gamma,
                            const float* __restrict__ beta,
                            const float* __restrict__ mean,
                            const float* __restrict__ var,
                            float* __restrict__ xout,
                            _Float16* __restrict__ xh, int n) {
  int i = blockIdx.x * blockDim.x + threadIdx.x;
  if (i >= n) return;
  int d = i & (DMODEL - 1);
  float s = x[i] + tadd[i];
  float y = gamma[d] * (s - mean[d]) * rsqrtf(var[d] + EPS) + beta[d];
  xout[i] = y;
  xh[i] = (_Float16)y;
}

// ---------------- host driver ---------------------------------------------

extern "C" void kernel_launch(void* const* d_in, const int* in_sizes, int n_in,
                              void* d_out, int out_size, void* d_ws, size_t ws_size,
                              hipStream_t stream) {
  const int*   seq  = (const int*)d_in[0];
  const float* emb  = (const float*)d_in[1];
  const float* pes  = (const float*)d_in[2];
  const float* wq   = (const float*)d_in[3];
  const float* bq   = (const float*)d_in[4];
  const float* wk   = (const float*)d_in[5];
  const float* bk   = (const float*)d_in[6];
  const float* wv   = (const float*)d_in[7];
  const float* bv   = (const float*)d_in[8];
  const float* wo   = (const float*)d_in[9];
  const float* bo   = (const float*)d_in[10];
  const float* w1   = (const float*)d_in[11];
  const float* b1   = (const float*)d_in[12];
  const float* w2   = (const float*)d_in[13];
  const float* b2   = (const float*)d_in[14];
  const float* ang  = (const float*)d_in[15];
  const float* anb  = (const float*)d_in[16];
  const float* anm  = (const float*)d_in[17];
  const float* anv  = (const float*)d_in[18];
  const float* fng  = (const float*)d_in[19];
  const float* fnb  = (const float*)d_in[20];
  const float* fnm  = (const float*)d_in[21];
  const float* fnv  = (const float*)d_in[22];

  // workspace layout (~320 MB total)
  char* ws = (char*)d_ws;
  size_t off = 0;
  auto alloc = [&](size_t bytes) -> void* {
    void* p = ws + off;
    off = (off + bytes + 255) & ~(size_t)255;
    return p;
  };
  float*     X     = (float*)alloc((size_t)MROWS * DMODEL * 4);
  _Float16*  Xh    = (_Float16*)alloc((size_t)MROWS * DMODEL * 2);
  _Float16*  Q     = (_Float16*)alloc((size_t)MROWS * DMODEL * 2);
  _Float16*  Kb    = (_Float16*)alloc((size_t)MROWS * DMODEL * 2);
  _Float16*  V     = (_Float16*)alloc((size_t)MROWS * DMODEL * 2);
  _Float16*  HEADS = (_Float16*)alloc((size_t)MROWS * DMODEL * 2);
  float*     TMP   = (float*)alloc((size_t)MROWS * DMODEL * 4);
  _Float16*  FFh   = (_Float16*)alloc((size_t)MROWS * DFF * 2);
  _Float16*  WQKVh = (_Float16*)alloc((size_t)NLAYER * DMODEL * 3 * DMODEL * 2);
  _Float16*  WOh   = (_Float16*)alloc((size_t)NLAYER * DMODEL * DMODEL * 2);
  _Float16*  W1h   = (_Float16*)alloc((size_t)NLAYER * DMODEL * DFF * 2);
  _Float16*  W2h   = (_Float16*)alloc((size_t)NLAYER * DFF * DMODEL * 2);
  float*     BQKV  = (float*)alloc((size_t)NLAYER * 3 * DMODEL * 4);

  const int TB = 256;
  // weight repack / f16 conversion (deterministic, done every call)
  {
    int tot = NLAYER * DMODEL * 3 * DMODEL;
    repack_qkv_w<<<(tot + TB - 1) / TB, TB, 0, stream>>>(wq, wk, wv, WQKVh, tot);
    int tb = NLAYER * 3 * DMODEL;
    repack_qkv_b<<<(tb + TB - 1) / TB, TB, 0, stream>>>(bq, bk, bv, BQKV, tb);
    int c1 = NLAYER * DMODEL * DMODEL;
    cvt_f32_f16<<<(c1 + TB - 1) / TB, TB, 0, stream>>>(wo, WOh, c1);
    int c2 = NLAYER * DMODEL * DFF;
    cvt_f32_f16<<<(c2 + TB - 1) / TB, TB, 0, stream>>>(w1, W1h, c2);
    cvt_f32_f16<<<(c2 + TB - 1) / TB, TB, 0, stream>>>(w2, W2h, c2);
  }

  // x = embedding[seq] + pes
  embed_kernel<<<(MROWS * DMODEL) / TB, TB, 0, stream>>>(seq, emb, pes, X, Xh);

  const dim3 blk(TB);
  for (int l = 0; l < NLAYER; ++l) {
    // QKV projection: [8192,1024] x [1024,3072] -> q,k,v [B,H,S,HD] f16
    gemm_wmma<0><<<dim3(3 * DMODEL / GBN, MROWS / GBM), blk, 0, stream>>>(
        Xh, WQKVh + (size_t)l * DMODEL * 3 * DMODEL, BQKV + (size_t)l * 3 * DMODEL,
        nullptr, nullptr, Q, Kb, V, 3 * DMODEL, DMODEL);

    // fused attention: 128 KB dynamic LDS score panel per block
    attn_fused<<<BATCH * NHEAD * (SEQ / 32), blk, 32 * SEQ * sizeof(float), stream>>>(
        Q, Kb, V, HEADS);

    // out-proj: heads @ wo + bo -> TMP (f32)
    gemm_wmma<1><<<dim3(DMODEL / GBN, MROWS / GBM), blk, 0, stream>>>(
        HEADS, WOh + (size_t)l * DMODEL * DMODEL, bo + (size_t)l * DMODEL,
        TMP, nullptr, nullptr, nullptr, nullptr, DMODEL, DMODEL);

    // x = BN(x + attn_out)
    bn_residual<<<(MROWS * DMODEL) / TB, TB, 0, stream>>>(
        X, TMP, ang + (size_t)l * DMODEL, anb + (size_t)l * DMODEL,
        anm + (size_t)l * DMODEL, anv + (size_t)l * DMODEL, X, Xh, MROWS * DMODEL);

    // ffn1: relu(x @ w1 + b1) -> FFh (f16)
    gemm_wmma<2><<<dim3(DFF / GBN, MROWS / GBM), blk, 0, stream>>>(
        Xh, W1h + (size_t)l * DMODEL * DFF, b1 + (size_t)l * DFF,
        nullptr, FFh, nullptr, nullptr, nullptr, DFF, DMODEL);

    // ffn2: FFh @ w2 + b2 -> TMP (f32)
    gemm_wmma<1><<<dim3(DMODEL / GBN, MROWS / GBM), blk, 0, stream>>>(
        FFh, W2h + (size_t)l * DFF * DMODEL, b2 + (size_t)l * DMODEL,
        TMP, nullptr, nullptr, nullptr, nullptr, DMODEL, DFF);

    // x = BN(x + ffn); last layer writes straight into d_out
    float* xdst = (l == NLAYER - 1) ? (float*)d_out : X;
    bn_residual<<<(MROWS * DMODEL) / TB, TB, 0, stream>>>(
        X, TMP, fng + (size_t)l * DMODEL, fnb + (size_t)l * DMODEL,
        fnm + (size_t)l * DMODEL, fnv + (size_t)l * DMODEL, xdst, Xh, MROWS * DMODEL);
  }
}
